// FlashAttentionCompatibleSparseAttention_22634477650390
// MI455X (gfx1250) — matrix operand order repaired and checked by reading, not compiled, not run on hardware
//
#include <hip/hip_runtime.h>
#include <hip/hip_bf16.h>
#include <stdint.h>

// ---------------------------------------------------------------------------
// Shapes (fixed by the reference): B=2, N=2048, C=512, H=8, D=64
// ---------------------------------------------------------------------------
#define B_  2
#define N_  2048
#define C_  512
#define H_  8
#define D_  64
#define SCALE_ 0.125f  // D^-0.5

typedef __bf16 bf16;
typedef __attribute__((ext_vector_type(16))) __bf16 v16bf;
typedef __attribute__((ext_vector_type(8)))  __bf16 v8bf;
typedef __attribute__((ext_vector_type(8)))  float  v8f;

__device__ __forceinline__ bf16 f2bf(float x) {
  union { float f; uint32_t u; } a; a.f = x;
  uint32_t r = a.u + 0x7FFFu + ((a.u >> 16) & 1u);  // round-to-nearest-even
  union { uint16_t s; bf16 b; } o; o.s = (uint16_t)(r >> 16);
  return o.b;
}

__device__ __forceinline__ v16bf mk16(v8bf lo, v8bf hi) {
  union { v16bf v; v8bf h[2]; } u; u.h[0] = lo; u.h[1] = hi; return u.v;
}

__device__ __forceinline__ v8f wmma_bf16(v16bf a, v16bf b, v8f c) {
  // D = A(16x32 bf16) * B(32x16 bf16) + C(16x16 f32)
  return __builtin_amdgcn_wmma_f32_16x16x32_bf16(false, a, false, b,
                                                 (short)0, c, false, false);
}

// ---------------------------------------------------------------------------
// Async global -> LDS copy, 16 bytes per lane (CDNA5 GLOBAL_LOAD_ASYNC_TO_LDS,
// tracked by ASYNCcnt). LDS offset = low 32 bits of the generic pointer.
// ---------------------------------------------------------------------------
__device__ __forceinline__ void async_copy16(void* lds_ptr, const void* gptr) {
  uint32_t lds_off = (uint32_t)(uintptr_t)lds_ptr;
  uint64_t gaddr   = (uint64_t)(uintptr_t)gptr;
  asm volatile("global_load_async_to_lds_b128 %0, %1, off"
               :: "v"(lds_off), "v"(gaddr)
               : "memory");
}
__device__ __forceinline__ void wait_async0() {
  asm volatile("s_wait_asynccnt 0" ::: "memory");
}

// ---------------------------------------------------------------------------
// fp32 -> bf16 conversion
// ---------------------------------------------------------------------------
__global__ void cvt_f32_bf16(const float* __restrict__ in, bf16* __restrict__ out, int n) {
  int i = blockIdx.x * blockDim.x + threadIdx.x;
  if (i < n) out[i] = f2bf(in[i]);
}

// ---------------------------------------------------------------------------
// QKV GEMM: qkv[m, j] = sum_k x[m,k] * w_qkv[j,k]   (M=4096, Nout=1536, K=512)
// Epilogue scatters into Q/K/V [B][H][N][D] bf16.
// Block = 256 threads (8 waves), block tile 64x64, K-step 32, double-buffered
// async global->LDS staging overlapping the WMMAs of the current step.
// ---------------------------------------------------------------------------
__global__ __launch_bounds__(256) void gemm_qkv(const bf16* __restrict__ A,
                                                const bf16* __restrict__ W,
                                                bf16* __restrict__ Qd,
                                                bf16* __restrict__ Kd,
                                                bf16* __restrict__ Vd) {
  __shared__ alignas(32) bf16 Al[2][64 * 40];
  __shared__ alignas(32) bf16 Bl[2][64 * 40];
  const int tid  = threadIdx.x;
  const int wave = tid >> 5, lane = tid & 31;
  const int l16  = lane & 15, half = lane >> 4;
  const int m0 = blockIdx.y * 64, n0 = blockIdx.x * 64;
  const int wm = (wave & 3) * 16, wn = (wave >> 2) * 32;
  const int r = tid >> 2, co = (tid & 3) * 8;

  auto stage = [&](int buf, int k0) {
    async_copy16(&Al[buf][r * 40 + co], A + (size_t)(m0 + r) * C_ + k0 + co);
    async_copy16(&Bl[buf][r * 40 + co], W + (size_t)(n0 + r) * C_ + k0 + co);
  };

  v8f acc0 = {}, acc1 = {};
  stage(0, 0);
  for (int s = 0; s < C_ / 32; ++s) {
    wait_async0();
    __syncthreads();
    const int cur = s & 1;
    if (s + 1 < C_ / 32) stage(cur ^ 1, (s + 1) * 32);  // overlap with WMMAs below
    const bf16* Ac = &Al[cur][0];
    const bf16* Bc = &Bl[cur][0];
    // A fragment: 16x32, lane holds row l16; k = 16*half + e
    v16bf af = mk16(*(const v8bf*)(Ac + (wm + l16) * 40 + 16 * half),
                    *(const v8bf*)(Ac + (wm + l16) * 40 + 16 * half + 8));
    // B fragment: 32x16, lane holds col l16; k = 16*half + e  (B[k][n] = W[j0+n][k])
    v16bf b0 = mk16(*(const v8bf*)(Bc + (wn + l16) * 40 + 16 * half),
                    *(const v8bf*)(Bc + (wn + l16) * 40 + 16 * half + 8));
    v16bf b1 = mk16(*(const v8bf*)(Bc + (wn + 16 + l16) * 40 + 16 * half),
                    *(const v8bf*)(Bc + (wn + 16 + l16) * 40 + 16 * half + 8));
    acc0 = wmma_bf16(af, b0, acc0);
    acc1 = wmma_bf16(af, b1, acc1);
  }
  // C layout: element rr <-> row m = rr + 8*half, col n = l16
#pragma unroll
  for (int rr = 0; rr < 8; ++rr) {
    int m = m0 + wm + rr + 8 * half;
    int b = m >> 11, nrow = m & (N_ - 1);
#pragma unroll
    for (int t = 0; t < 2; ++t) {
      int j = n0 + wn + t * 16 + l16;
      float val = (t == 0) ? acc0[rr] : acc1[rr];
      int three = j >> 9, hh = (j >> 6) & 7, d = j & 63;
      bf16* dst = (three == 0) ? Qd : (three == 1) ? Kd : Vd;
      dst[(((size_t)b * H_ + hh) * N_ + nrow) * D_ + d] = f2bf(val);
    }
  }
}

// ---------------------------------------------------------------------------
// Flash attention, one (b,h) per blockIdx.y, 64 query rows per block,
// 4 waves each owning 16 query rows. Keys processed in chunks of 32.
// K chunk staged via double-buffered async global->LDS; V chunk transposed
// through VGPRs into a double-buffered LDS tile.
// ---------------------------------------------------------------------------
__global__ __launch_bounds__(128) void attn(const bf16* __restrict__ Q,
                                            const bf16* __restrict__ Kk,
                                            const bf16* __restrict__ V,
                                            const unsigned char* __restrict__ mask,
                                            bf16* __restrict__ O) {
  __shared__ alignas(32) bf16 Kl[2][32 * 80];  // K chunk  [key][d], stride 80
  __shared__ alignas(32) bf16 VT[2][64 * 40];  // V chunk transposed [d][key], stride 40
  __shared__ alignas(32) bf16 Ps[4][16 * 32];  // per-wave P bounce buffer

  const int tid  = threadIdx.x;
  const int wave = tid >> 5, lane = tid & 31;
  const int l16  = lane & 15, half = lane >> 4;
  const int bh   = blockIdx.y;
  const bf16* Qb = Q  + (size_t)bh * N_ * D_;
  const bf16* Kb = Kk + (size_t)bh * N_ * D_;
  const bf16* Vb = V  + (size_t)bh * N_ * D_;
  const int q0 = blockIdx.x * 64 + wave * 16;

  auto stageK = [&](int buf, int key0) {
#pragma unroll
    for (int c = tid; c < 256; c += 128) {
      int ky = c >> 3, of = (c & 7) * 8;
      async_copy16(&Kl[buf][ky * 80 + of], Kb + (size_t)(key0 + ky) * D_ + of);
    }
  };
  auto stageV = [&](int buf, int key0) {
    int ky = tid >> 2, dof = (tid & 3) * 16;
    v8bf v0 = *(const v8bf*)(Vb + (size_t)(key0 + ky) * D_ + dof);
    v8bf v1 = *(const v8bf*)(Vb + (size_t)(key0 + ky) * D_ + dof + 8);
#pragma unroll
    for (int i = 0; i < 8; ++i) {
      VT[buf][(dof + i) * 40 + ky]     = v0[i];
      VT[buf][(dof + 8 + i) * 40 + ky] = v1[i];
    }
  };

  // Q fragments for the two 32-wide d halves (A layout, contiguous 16B chunks)
  v16bf qf0 = mk16(*(const v8bf*)(Qb + (size_t)(q0 + l16) * D_ + 16 * half),
                   *(const v8bf*)(Qb + (size_t)(q0 + l16) * D_ + 16 * half + 8));
  v16bf qf1 = mk16(*(const v8bf*)(Qb + (size_t)(q0 + l16) * D_ + 32 + 16 * half),
                   *(const v8bf*)(Qb + (size_t)(q0 + l16) * D_ + 32 + 16 * half + 8));

  const float NEG = -__builtin_inff();
  float rm[8], rs[8];
#pragma unroll
  for (int i = 0; i < 8; ++i) { rm[i] = NEG; rs[i] = 0.f; }
  v8f acc0 = {}, acc1 = {}, acc2 = {}, acc3 = {};

  stageK(0, 0);
  stageV(0, 0);
  for (int s = 0; s < N_ / 32; ++s) {
    const int key0 = s * 32;
    const int cur  = s & 1;
    wait_async0();
    __syncthreads();  // publishes Kl[cur] (async) and VT[cur] (ds stores)
    if (s + 1 < N_ / 32) {   // overlap next chunk's staging with this chunk's math
      stageK(cur ^ 1, key0 + 32);
      stageV(cur ^ 1, key0 + 32);
    }
    const bf16* Kc = &Kl[cur][0];
    const bf16* Vc = &VT[cur][0];

    // --- S = Q K^T for two 16-key subtiles (contract over d=64 in two WMMAs) ---
    v8f s0 = {}, s1 = {};
#pragma unroll
    for (int dh = 0; dh < 2; ++dh) {
      v16bf qa = (dh == 0) ? qf0 : qf1;
      v16bf kb0 = mk16(*(const v8bf*)(Kc + l16 * 80 + 32 * dh + 16 * half),
                       *(const v8bf*)(Kc + l16 * 80 + 32 * dh + 16 * half + 8));
      v16bf kb1 = mk16(*(const v8bf*)(Kc + (16 + l16) * 80 + 32 * dh + 16 * half),
                       *(const v8bf*)(Kc + (16 + l16) * 80 + 32 * dh + 16 * half + 8));
      s0 = wmma_bf16(qa, kb0, s0);
      s1 = wmma_bf16(qa, kb1, s1);
    }

    // --- mask + online softmax (row stats replicated across each 16-lane half) ---
    float p0[8], p1[8], corr[8];
#pragma unroll
    for (int rr = 0; rr < 8; ++rr) {
      int gm = q0 + rr + 8 * half;
      float a0 = mask[(size_t)gm * N_ + key0 + l16]      ? s0[rr] * SCALE_ : NEG;
      float a1 = mask[(size_t)gm * N_ + key0 + 16 + l16] ? s1[rr] * SCALE_ : NEG;
      float v = fmaxf(a0, a1);
#pragma unroll
      for (int o = 1; o < 16; o <<= 1) v = fmaxf(v, __shfl_xor(v, o, 32));
      float nm = fmaxf(rm[rr], v);
      if (nm == NEG) { corr[rr] = 1.f; p0[rr] = 0.f; p1[rr] = 0.f; }
      else {
        corr[rr] = __expf(rm[rr] - nm);
        p0[rr] = __expf(a0 - nm);
        p1[rr] = __expf(a1 - nm);
      }
      float sump = p0[rr] + p1[rr];
#pragma unroll
      for (int o = 1; o < 16; o <<= 1) sump += __shfl_xor(sump, o, 32);
      rs[rr] = rs[rr] * corr[rr] + sump;
      rm[rr] = nm;
      acc0[rr] *= corr[rr]; acc1[rr] *= corr[rr];
      acc2[rr] *= corr[rr]; acc3[rr] *= corr[rr];
    }

    // --- bounce P (C layout -> A layout) through per-wave LDS scratch ---
    bf16* ps = &Ps[wave][0];
#pragma unroll
    for (int rr = 0; rr < 8; ++rr) {
      ps[(rr + 8 * half) * 32 + l16]      = f2bf(p0[rr]);
      ps[(rr + 8 * half) * 32 + 16 + l16] = f2bf(p1[rr]);
    }
    __syncthreads();
    v16bf pf = mk16(*(const v8bf*)(ps + l16 * 32 + 16 * half),
                    *(const v8bf*)(ps + l16 * 32 + 16 * half + 8));
    // --- acc += P V  (B[k=key][n=d] = VT[d][key], contiguous in key) ---
#pragma unroll
    for (int dj = 0; dj < 4; ++dj) {
      v16bf vb = mk16(*(const v8bf*)(Vc + (dj * 16 + l16) * 40 + 16 * half),
                      *(const v8bf*)(Vc + (dj * 16 + l16) * 40 + 16 * half + 8));
      if      (dj == 0) acc0 = wmma_bf16(pf, vb, acc0);
      else if (dj == 1) acc1 = wmma_bf16(pf, vb, acc1);
      else if (dj == 2) acc2 = wmma_bf16(pf, vb, acc2);
      else              acc3 = wmma_bf16(pf, vb, acc3);
    }
  }

  // --- normalize and store as bf16 [B][N][C] with column h*64+d ---
  const int b = bh >> 3, hh = bh & 7;
#pragma unroll
  for (int rr = 0; rr < 8; ++rr) {
    int gm = q0 + rr + 8 * half;
    float inv = rs[rr] > 0.f ? 1.f / rs[rr] : 0.f;
    size_t base = ((size_t)b * N_ + gm) * C_ + hh * D_;
    O[base + 0 * 16 + l16] = f2bf(acc0[rr] * inv);
    O[base + 1 * 16 + l16] = f2bf(acc1[rr] * inv);
    O[base + 2 * 16 + l16] = f2bf(acc2[rr] * inv);
    O[base + 3 * 16 + l16] = f2bf(acc3[rr] * inv);
  }
}

// ---------------------------------------------------------------------------
// Output projection: out[m,j] = sum_k attn[m,k] * w_proj[j,k] + b_proj[j]
// Same double-buffered tiling as gemm_qkv; fp32 output.
// ---------------------------------------------------------------------------
__global__ __launch_bounds__(256) void gemm_proj(const bf16* __restrict__ A,
                                                 const bf16* __restrict__ W,
                                                 const float* __restrict__ bias,
                                                 float* __restrict__ Out) {
  __shared__ alignas(32) bf16 Al[2][64 * 40];
  __shared__ alignas(32) bf16 Bl[2][64 * 40];
  const int tid  = threadIdx.x;
  const int wave = tid >> 5, lane = tid & 31;
  const int l16  = lane & 15, half = lane >> 4;
  const int m0 = blockIdx.y * 64, n0 = blockIdx.x * 64;
  const int wm = (wave & 3) * 16, wn = (wave >> 2) * 32;
  const int r = tid >> 2, co = (tid & 3) * 8;

  auto stage = [&](int buf, int k0) {
    async_copy16(&Al[buf][r * 40 + co], A + (size_t)(m0 + r) * C_ + k0 + co);
    async_copy16(&Bl[buf][r * 40 + co], W + (size_t)(n0 + r) * C_ + k0 + co);
  };

  v8f acc0 = {}, acc1 = {};
  stage(0, 0);
  for (int s = 0; s < C_ / 32; ++s) {
    wait_async0();
    __syncthreads();
    const int cur = s & 1;
    if (s + 1 < C_ / 32) stage(cur ^ 1, (s + 1) * 32);
    const bf16* Ac = &Al[cur][0];
    const bf16* Bc = &Bl[cur][0];
    v16bf af = mk16(*(const v8bf*)(Ac + (wm + l16) * 40 + 16 * half),
                    *(const v8bf*)(Ac + (wm + l16) * 40 + 16 * half + 8));
    v16bf b0 = mk16(*(const v8bf*)(Bc + (wn + l16) * 40 + 16 * half),
                    *(const v8bf*)(Bc + (wn + l16) * 40 + 16 * half + 8));
    v16bf b1 = mk16(*(const v8bf*)(Bc + (wn + 16 + l16) * 40 + 16 * half),
                    *(const v8bf*)(Bc + (wn + 16 + l16) * 40 + 16 * half + 8));
    acc0 = wmma_bf16(af, b0, acc0);
    acc1 = wmma_bf16(af, b1, acc1);
  }
#pragma unroll
  for (int rr = 0; rr < 8; ++rr) {
    int m = m0 + wm + rr + 8 * half;
#pragma unroll
    for (int t = 0; t < 2; ++t) {
      int j = n0 + wn + t * 16 + l16;
      float val = (t == 0) ? acc0[rr] : acc1[rr];
      Out[(size_t)m * C_ + j] = val + bias[j];
    }
  }
}

// ---------------------------------------------------------------------------
// Host-side launcher
// ---------------------------------------------------------------------------
extern "C" void kernel_launch(void* const* d_in, const int* in_sizes, int n_in,
                              void* d_out, int out_size, void* d_ws, size_t ws_size,
                              hipStream_t stream) {
  (void)in_sizes; (void)n_in; (void)out_size; (void)ws_size;
  const float*         x      = (const float*)d_in[0];
  const unsigned char* mask   = (const unsigned char*)d_in[1];  // jnp bool_ = 1 byte
  const float*         w_qkv  = (const float*)d_in[2];
  const float*         w_proj = (const float*)d_in[3];
  const float*         b_proj = (const float*)d_in[4];
  float*               out    = (float*)d_out;

  char* ws = (char*)d_ws;
  bf16* x_bf     = (bf16*)(ws + 0x0000000);  // 4 MB  (4096 x 512)
  bf16* wqkv_bf  = (bf16*)(ws + 0x0400000);  // 1.5 MB (1536 x 512)
  bf16* wproj_bf = (bf16*)(ws + 0x0600000);  // 0.5 MB (512 x 512)
  bf16* Qd       = (bf16*)(ws + 0x0700000);  // 4 MB  [B][H][N][D]
  bf16* Kd       = (bf16*)(ws + 0x0B00000);  // 4 MB
  bf16* Vd       = (bf16*)(ws + 0x0F00000);  // 4 MB
  bf16* attn_bf  = (bf16*)(ws + 0x1300000);  // 4 MB  (4096 x 512)

  const int n_x  = B_ * N_ * C_;       // 2,097,152
  const int n_wq = 3 * C_ * C_;        //   786,432
  const int n_wp = C_ * C_;            //   262,144

  cvt_f32_bf16<<<(n_x  + 255) / 256, 256, 0, stream>>>(x,      x_bf,     n_x);
  cvt_f32_bf16<<<(n_wq + 255) / 256, 256, 0, stream>>>(w_qkv,  wqkv_bf,  n_wq);
  cvt_f32_bf16<<<(n_wp + 255) / 256, 256, 0, stream>>>(w_proj, wproj_bf, n_wp);

  dim3 g1(3 * C_ / 64, (B_ * N_) / 64);   // (24, 64)
  gemm_qkv<<<g1, 256, 0, stream>>>(x_bf, wqkv_bf, Qd, Kd, Vd);

  dim3 g2(N_ / 64, B_ * H_);              // (32, 16)
  attn<<<g2, 128, 0, stream>>>(Qd, Kd, Vd, mask, attn_bf);

  dim3 g3(C_ / 64, (B_ * N_) / 64);       // (8, 64)
  gemm_proj<<<g3, 256, 0, stream>>>(attn_bf, wproj_bf, b_proj, out);
}
